// DeepseekV3Attention_62989990363215
// MI455X (gfx1250) — matrix operand order repaired and checked by reading, not compiled, not run on hardware
//
#include <hip/hip_runtime.h>
#include <hip/hip_bf16.h>

typedef unsigned short ushort_t;
typedef __attribute__((ext_vector_type(16))) __bf16 v16bf;
typedef __attribute__((ext_vector_type(8)))  float  v8f;
typedef __attribute__((ext_vector_type(4)))  float  f32x4;

#define WMMA_BF16(a, b, c) \
  __builtin_amdgcn_wmma_f32_16x16x32_bf16(false, (a), false, (b), (short)0, (c), false, false)

// ---------------------------------------------------------------- constants
constexpr int SEQ   = 2048;
constexpr int HID   = 4096;
constexpr int NH    = 32;
constexpr int QLORA = 1536;
constexpr int KVLORA= 512;
constexpr int ROPE  = 64;
constexpr int NOPE  = 128;
constexpr int VDIM  = 128;
constexpr int QHD   = NOPE + ROPE;          // 192
constexpr float SCALING = 0.07216878364870323f;   // 192^-0.5
constexpr float EPSV = 1e-6f;

// ---------------------------------------------------------------- helpers
static __device__ inline ushort_t f2bf(float f) {
  unsigned int u = __float_as_uint(f);
  u += 0x7FFFu + ((u >> 16) & 1u);          // round-to-nearest-even
  return (ushort_t)(u >> 16);
}

union Frag { v16bf v; f32x4 f[2]; };

// generic->LDS byte offset (flat LDS aperture keeps offset in addr[31:0])
static __device__ inline unsigned lds_off(const void* p) {
  return (unsigned)(unsigned long long)p;
}

// async copy 32B (two b128 beats) global -> LDS; tracked by ASYNCcnt
static __device__ inline void async_cp32(unsigned lds, const ushort_t* g) {
  asm volatile("global_load_async_to_lds_b128 %0, %1, off\n\t"
               "global_load_async_to_lds_b128 %0, %1, off offset:16"
               :: "v"(lds), "v"(g) : "memory");
}
static __device__ inline void wait_async0() {
  asm volatile("s_wait_asynccnt 0x0" ::: "memory");
}

// xor-shuffle within 16-lane groups via ds_swizzle (and=0x1F, xor=M keeps halves separate)
template <int M>
static __device__ inline float swz(float v) {
  return __int_as_float(__builtin_amdgcn_ds_swizzle(__float_as_int(v), (M << 10) | 0x1F));
}
static __device__ inline float rmax16(float v) {
  v = fmaxf(v, swz<1>(v)); v = fmaxf(v, swz<2>(v));
  v = fmaxf(v, swz<4>(v)); v = fmaxf(v, swz<8>(v));
  return v;
}
static __device__ inline float rsum16(float v) {
  v += swz<1>(v); v += swz<2>(v); v += swz<4>(v); v += swz<8>(v);
  return v;
}

// ---------------------------------------------------------------- fp32 -> bf16 copy
__global__ __launch_bounds__(256) void cvt_bf16(const float* __restrict__ in,
                                                ushort_t* __restrict__ out, long n) {
  for (long i = (long)blockIdx.x * blockDim.x + threadIdx.x; i < n;
       i += (long)gridDim.x * blockDim.x)
    out[i] = f2bf(in[i]);
}

// ---------------------------------------------------------------- GEMM: C[M,N](f32) = A[M,K](bf16) * W[N,K](bf16)^T
// block = 128 threads (4 waves); block tile 64(M) x 64(N); wave w -> rows [w*16, w*16+16).
// B tile (64 cols x 32 k) staged per-block into LDS with double-buffered
// global_load_async_to_lds_b128 (ASYNCcnt), shared by all 4 waves.
// All 4 B-frags are loaded into distinct registers before the 4 WMMAs so a
// single s_wait_dscnt covers the whole K-step.
__global__ __launch_bounds__(128) void gemm_bf16_wmma(const ushort_t* __restrict__ A,
                                                      const ushort_t* __restrict__ W,
                                                      float* __restrict__ C,
                                                      int M, int N, int K) {
  __shared__ ushort_t Bt[2][64 * 32];       // [buf][col][k] bf16, 4KB each

  const int tid  = threadIdx.x;
  const int lane = tid & 31;
  const int wv   = tid >> 5;
  const int r    = lane & 15;
  const int hf   = lane >> 4;
  const int row0 = blockIdx.y * 64 + wv * 16;
  const int col0 = blockIdx.x * 64;

  const ushort_t* arow = A + (size_t)(row0 + r) * K;

  // async-loader assignment: thread tid copies 32B: col c, k-segment seg*16
  const int c   = tid >> 1;
  const int seg = tid & 1;
  const ushort_t* gsrc = W + (size_t)(col0 + c) * K + seg * 16;
  const unsigned lbuf0 = lds_off(&Bt[0][c * 32 + seg * 16]);
  const unsigned lbuf1 = lds_off(&Bt[1][c * 32 + seg * 16]);

  v8f acc[4];
#pragma unroll
  for (int t = 0; t < 4; ++t)
#pragma unroll
    for (int j = 0; j < 8; ++j) acc[t][j] = 0.0f;

  const int nsteps = K >> 5;
  async_cp32(lbuf0, gsrc);                  // prologue: stage 0 in flight

  for (int s = 0; s < nsteps; ++s) {
    const int kb = s * 32;
    wait_async0();                          // my beats of current stage landed
    __syncthreads();                        // whole tile visible; prev readers done

    // branchless next-stage issue: last step re-copies stage 0 into the dead
    // buffer (never read; retired by endpgm's implicit wait-idle)
    const int knext = (s + 1 < nsteps) ? kb + 32 : 0;
    async_cp32((s & 1) ? lbuf0 : lbuf1, gsrc + knext);
    if (kb + 32 < K)
      __builtin_prefetch(arow + kb + 32, 0, 1);

    Frag a;
    a.f[0] = *(const f32x4*)(arow + kb + hf * 16);
    a.f[1] = *(const f32x4*)(arow + kb + hf * 16 + 8);

    const ushort_t* bb = &Bt[s & 1][0];
    Frag b[4];
#pragma unroll
    for (int t = 0; t < 4; ++t) {
      b[t].f[0] = *(const f32x4*)(bb + (t * 16 + r) * 32 + hf * 16);
      b[t].f[1] = *(const f32x4*)(bb + (t * 16 + r) * 32 + hf * 16 + 8);
    }
#pragma unroll
    for (int t = 0; t < 4; ++t)
      acc[t] = WMMA_BF16(a.v, b[t].v, acc[t]);
  }

#pragma unroll
  for (int t = 0; t < 4; ++t)
#pragma unroll
    for (int j = 0; j < 8; ++j) {
      int row = row0 + hf * 8 + j;
      int col = col0 + t * 16 + r;
      C[(size_t)row * N + col] = acc[t][j];
    }
}

// ---------------------------------------------------------------- RMSNorm (f32 in, bf16 out)
__global__ __launch_bounds__(256) void rmsnorm_bf16(const float* __restrict__ X,
                                                    const float* __restrict__ w,
                                                    ushort_t* __restrict__ out,
                                                    int D, int inStride, int outStride) {
  __shared__ float red[256];
  const int row = blockIdx.x, tid = threadIdx.x;
  const float* x = X + (size_t)row * inStride;
  float ss = 0.0f;
  for (int i = tid; i < D; i += 256) ss += x[i] * x[i];
  red[tid] = ss;
  __syncthreads();
  for (int s = 128; s > 0; s >>= 1) {
    if (tid < s) red[tid] += red[tid + s];
    __syncthreads();
  }
  const float rs = rsqrtf(red[0] / (float)D + EPSV);
  for (int i = tid; i < D; i += 256)
    out[(size_t)row * outStride + i] = f2bf(x[i] * rs * w[i]);
}

// ---------------------------------------------------------------- build Q: rope + head-major + scaling
__global__ __launch_bounds__(192) void build_q(const float* __restrict__ q,
                                               const float* __restrict__ cs,
                                               const float* __restrict__ sn,
                                               ushort_t* __restrict__ Qa) {
  const int s = blockIdx.x, h = blockIdx.y, d = threadIdx.x;
  const float* qr = q + (size_t)s * (NH * QHD) + h * QHD;
  float val;
  if (d < ROPE) {
    float x = qr[NOPE + d];
    float p = (d < 32) ? -qr[NOPE + d + 32] : qr[NOPE + d - 32];
    val = x * cs[(size_t)s * ROPE + d] + p * sn[(size_t)s * ROPE + d];
  } else {
    val = qr[d - ROPE];
  }
  Qa[((size_t)h * SEQ + s) * QHD + d] = f2bf(val * SCALING);
}

// ---------------------------------------------------------------- build K (rope|nope) and V (d-major transpose)
__global__ __launch_bounds__(320) void build_kv(const float* __restrict__ ckv,
                                                const float* __restrict__ kvb,
                                                const float* __restrict__ cs,
                                                const float* __restrict__ sn,
                                                ushort_t* __restrict__ Ka,
                                                ushort_t* __restrict__ Vt) {
  const int s = blockIdx.x, h = blockIdx.y, t = threadIdx.x;
  if (t < QHD) {
    const int d = t;
    float val;
    if (d < ROPE) {
      float x = ckv[(size_t)s * (KVLORA + ROPE) + KVLORA + d];
      float p = (d < 32) ? -ckv[(size_t)s * (KVLORA + ROPE) + KVLORA + d + 32]
                         :  ckv[(size_t)s * (KVLORA + ROPE) + KVLORA + d - 32];
      val = x * cs[(size_t)s * ROPE + d] + p * sn[(size_t)s * ROPE + d];
    } else {
      val = kvb[(size_t)s * (NH * (NOPE + VDIM)) + h * (NOPE + VDIM) + (d - ROPE)];
    }
    Ka[((size_t)h * SEQ + s) * QHD + d] = f2bf(val);
  } else {
    const int d = t - QHD;                  // 0..127
    Vt[((size_t)h * VDIM + d) * SEQ + s] =
        f2bf(kvb[(size_t)s * (NH * (NOPE + VDIM)) + h * (NOPE + VDIM) + NOPE + d]);
  }
}

// ---------------------------------------------------------------- causal flash attention, 1 wave per (head, 16-query tile)
__global__ __launch_bounds__(32) void mla_flash(const ushort_t* __restrict__ Q,
                                                const ushort_t* __restrict__ Kc,
                                                const ushort_t* __restrict__ Vt,
                                                ushort_t* __restrict__ Obf) {
  __shared__ ushort_t P[16 * 32];           // score tile staged for A-frag reshape
  const int lane = threadIdx.x;
  const int r = lane & 15, hf = lane >> 4;
  const int qt = blockIdx.x, h = blockIdx.y;
  const int qbase = qt * 16;

  const ushort_t* qrow = Q + ((size_t)h * SEQ + qbase + r) * QHD;
  Frag qa[6];
#pragma unroll
  for (int dc = 0; dc < 6; ++dc) {
    qa[dc].f[0] = *(const f32x4*)(qrow + dc * 32 + hf * 16);
    qa[dc].f[1] = *(const f32x4*)(qrow + dc * 32 + hf * 16 + 8);
  }

  v8f O[8];
  float mrow[8], lrow[8];
#pragma unroll
  for (int t = 0; t < 8; ++t) {
#pragma unroll
    for (int j = 0; j < 8; ++j) O[t][j] = 0.0f;
    mrow[t] = -3.0e38f;
    lrow[t] = 0.0f;
  }

  const ushort_t* vbase = Vt + (size_t)h * VDIM * SEQ;  // [d][s]
  const int nch = (qbase + 15) / 32 + 1;

  for (int c = 0; c < nch; ++c) {
    const int kb = c * 32;
    v8f S0, S1;
#pragma unroll
    for (int j = 0; j < 8; ++j) { S0[j] = 0.0f; S1[j] = 0.0f; }

    const ushort_t* k0 = Kc + ((size_t)h * SEQ + kb + r) * QHD;
    const ushort_t* k1 = k0 + 16 * QHD;

    // QK^T: software-pipelined K-fragment loads (double-buffered), so each
    // WMMA waits only on loads issued one d-chunk earlier.
    Frag b0[2], b1[2];
    b0[0].f[0] = *(const f32x4*)(k0 + hf * 16);
    b0[0].f[1] = *(const f32x4*)(k0 + hf * 16 + 8);
    b1[0].f[0] = *(const f32x4*)(k1 + hf * 16);
    b1[0].f[1] = *(const f32x4*)(k1 + hf * 16 + 8);
#pragma unroll
    for (int dc = 0; dc < 6; ++dc) {
      const int cur = dc & 1, nxt = cur ^ 1;
      if (dc < 5) {
        b0[nxt].f[0] = *(const f32x4*)(k0 + (dc + 1) * 32 + hf * 16);
        b0[nxt].f[1] = *(const f32x4*)(k0 + (dc + 1) * 32 + hf * 16 + 8);
        b1[nxt].f[0] = *(const f32x4*)(k1 + (dc + 1) * 32 + hf * 16);
        b1[nxt].f[1] = *(const f32x4*)(k1 + (dc + 1) * 32 + hf * 16 + 8);
      }
      S0 = WMMA_BF16(qa[dc].v, b0[cur].v, S0);
      S1 = WMMA_BF16(qa[dc].v, b1[cur].v, S1);
    }

    float alpha[8];
#pragma unroll
    for (int j = 0; j < 8; ++j) {
      const int rowj = j + hf * 8;
      const int qpos = qbase + rowj;
      float s0 = (kb + r      > qpos) ? -3.0e38f : S0[j];
      float s1 = (kb + 16 + r > qpos) ? -3.0e38f : S1[j];
      float mx = rmax16(fmaxf(s0, s1));
      float nm = fmaxf(mrow[j], mx);
      float al = __expf(mrow[j] - nm);
      float p0 = __expf(s0 - nm);
      float p1 = __expf(s1 - nm);
      float rs = rsum16(p0 + p1);
      lrow[j] = lrow[j] * al + rs;
      mrow[j] = nm;
      alpha[j] = al;
      P[rowj * 32 + r]      = f2bf(p0);
      P[rowj * 32 + 16 + r] = f2bf(p1);
    }
#pragma unroll
    for (int t = 0; t < 8; ++t)
#pragma unroll
      for (int j = 0; j < 8; ++j) O[t][j] *= alpha[j];

    asm volatile("s_wait_dscnt 0x0" ::: "memory");

    Frag pa;
    pa.f[0] = *(const f32x4*)(&P[r * 32 + hf * 16]);
    pa.f[1] = *(const f32x4*)(&P[r * 32 + hf * 16 + 8]);

    // PV: double-buffered V-fragment loads
    Frag vb[2];
    {
      const ushort_t* vp0 = vbase + (size_t)r * SEQ + kb + hf * 16;
      vb[0].f[0] = *(const f32x4*)(vp0);
      vb[0].f[1] = *(const f32x4*)(vp0 + 8);
    }
#pragma unroll
    for (int vt = 0; vt < 8; ++vt) {
      const int cur = vt & 1, nxt = cur ^ 1;
      if (vt < 7) {
        const ushort_t* vp = vbase + (size_t)((vt + 1) * 16 + r) * SEQ + kb + hf * 16;
        vb[nxt].f[0] = *(const f32x4*)(vp);
        vb[nxt].f[1] = *(const f32x4*)(vp + 8);
      }
      O[vt] = WMMA_BF16(pa.v, vb[cur].v, O[vt]);
    }
  }

#pragma unroll
  for (int j = 0; j < 8; ++j) {
    const float inv = 1.0f / lrow[j];
    const size_t o = (size_t)(qbase + j + hf * 8) * (NH * VDIM) + (size_t)h * VDIM;
#pragma unroll
    for (int vt = 0; vt < 8; ++vt)
      Obf[o + vt * 16 + r] = f2bf(O[vt][j] * inv);
  }
}

// ---------------------------------------------------------------- host launch
extern "C" void kernel_launch(void* const* d_in, const int* in_sizes, int n_in,
                              void* d_out, int out_size, void* d_ws, size_t ws_size,
                              hipStream_t stream) {
  (void)in_sizes; (void)n_in; (void)out_size; (void)ws_size;

  const float* hs    = (const float*)d_in[0];   // (1,2048,4096)
  const float* cosp  = (const float*)d_in[1];   // (1,2048,64)
  const float* sinp  = (const float*)d_in[2];
  const float* w_q_a = (const float*)d_in[3];   // (1536,4096)
  const float* qlnw  = (const float*)d_in[4];   // (1536,)
  const float* w_q_b = (const float*)d_in[5];   // (6144,1536)
  const float* w_kva = (const float*)d_in[6];   // (576,4096)
  const float* kvlnw = (const float*)d_in[7];   // (512,)
  const float* w_kvb = (const float*)d_in[8];   // (8192,512)
  const float* w_o   = (const float*)d_in[9];   // (4096,4096)

  size_t off = 0;
  auto alloc = [&](size_t bytes) -> void* {
    size_t cur = (off + 255) & ~(size_t)255;
    off = cur + bytes;
    return (void*)((char*)d_ws + cur);
  };

  ushort_t* hs_bf   = (ushort_t*)alloc((size_t)SEQ * HID * 2);
  ushort_t* wqa_bf  = (ushort_t*)alloc((size_t)QLORA * HID * 2);
  ushort_t* wqb_bf  = (ushort_t*)alloc((size_t)NH * QHD * QLORA * 2);
  ushort_t* wkva_bf = (ushort_t*)alloc((size_t)(KVLORA + ROPE) * HID * 2);
  ushort_t* wkvb_bf = (ushort_t*)alloc((size_t)NH * (NOPE + VDIM) * KVLORA * 2);
  ushort_t* wo_bf   = (ushort_t*)alloc((size_t)HID * (NH * VDIM) * 2);
  float*    qlat_f  = (float*)   alloc((size_t)SEQ * QLORA * 4);
  ushort_t* qlat_bf = (ushort_t*)alloc((size_t)SEQ * QLORA * 2);
  float*    ckv_f   = (float*)   alloc((size_t)SEQ * (KVLORA + ROPE) * 4);
  ushort_t* klat_bf = (ushort_t*)alloc((size_t)SEQ * KVLORA * 2);
  float*    q_f     = (float*)   alloc((size_t)SEQ * NH * QHD * 4);
  float*    kv_f    = (float*)   alloc((size_t)SEQ * NH * (NOPE + VDIM) * 4);
  ushort_t* Qattn   = (ushort_t*)alloc((size_t)NH * SEQ * QHD * 2);
  ushort_t* Kattn   = (ushort_t*)alloc((size_t)NH * SEQ * QHD * 2);
  ushort_t* Vattn   = (ushort_t*)alloc((size_t)NH * VDIM * SEQ * 2);
  ushort_t* attn_bf = (ushort_t*)alloc((size_t)SEQ * NH * VDIM * 2);

  // 1) bf16 casts of streaming operands
  cvt_bf16<<<2048, 256, 0, stream>>>(hs,    hs_bf,   (long)SEQ * HID);
  cvt_bf16<<<2048, 256, 0, stream>>>(w_q_a, wqa_bf,  (long)QLORA * HID);
  cvt_bf16<<<2048, 256, 0, stream>>>(w_q_b, wqb_bf,  (long)NH * QHD * QLORA);
  cvt_bf16<<<2048, 256, 0, stream>>>(w_kva, wkva_bf, (long)(KVLORA + ROPE) * HID);
  cvt_bf16<<<2048, 256, 0, stream>>>(w_kvb, wkvb_bf, (long)NH * (NOPE + VDIM) * KVLORA);
  cvt_bf16<<<2048, 256, 0, stream>>>(w_o,   wo_bf,   (long)HID * NH * VDIM);

  // 2) down-projections
  gemm_bf16_wmma<<<dim3(QLORA / 64, SEQ / 64), 128, 0, stream>>>(
      hs_bf, wqa_bf, qlat_f, SEQ, QLORA, HID);
  gemm_bf16_wmma<<<dim3((KVLORA + ROPE) / 64, SEQ / 64), 128, 0, stream>>>(
      hs_bf, wkva_bf, ckv_f, SEQ, KVLORA + ROPE, HID);

  // 3) RMSNorm latents -> bf16
  rmsnorm_bf16<<<SEQ, 256, 0, stream>>>(qlat_f, qlnw, qlat_bf, QLORA, QLORA, QLORA);
  rmsnorm_bf16<<<SEQ, 256, 0, stream>>>(ckv_f, kvlnw, klat_bf, KVLORA, KVLORA + ROPE, KVLORA);

  // 4) up-projections
  gemm_bf16_wmma<<<dim3(NH * QHD / 64, SEQ / 64), 128, 0, stream>>>(
      qlat_bf, wqb_bf, q_f, SEQ, NH * QHD, QLORA);
  gemm_bf16_wmma<<<dim3(NH * (NOPE + VDIM) / 64, SEQ / 64), 128, 0, stream>>>(
      klat_bf, wkvb_bf, kv_f, SEQ, NH * (NOPE + VDIM), KVLORA);

  // 5) RoPE + head-major repack (+ score scaling folded into Q), V transposed d-major
  build_q<<<dim3(SEQ, NH), QHD, 0, stream>>>(q_f, cosp, sinp, Qattn);
  build_kv<<<dim3(SEQ, NH), QHD + VDIM, 0, stream>>>(ckv_f, kv_f, cosp, sinp, Kattn, Vattn);

  // 6) causal flash attention
  mla_flash<<<dim3(SEQ / 16, NH), 32, 0, stream>>>(Qattn, Kattn, Vattn, attn_bf);

  // 7) output projection -> fp32 d_out
  gemm_bf16_wmma<<<dim3(HID / 64, SEQ / 64), 128, 0, stream>>>(
      attn_bf, wo_bf, (float*)d_out, SEQ, HID, NH * VDIM);
}
